// TransformerLayer_6708738916836
// MI455X (gfx1250) — matrix operand order repaired and checked by reading, not compiled, not run on hardware
//
#include <hip/hip_runtime.h>
#include <hip/hip_bf16.h>

// Problem constants (B, C, L, D, H from reference)
#define B_  2
#define C_  256
#define L_  2048
#define D_  64
#define H_  8
#define HD_ 512
#define EPS_ 1e-8f

typedef __attribute__((ext_vector_type(16))) __bf16 v16bf;
typedef __attribute__((ext_vector_type(8)))  float  v8f;

__device__ __forceinline__ unsigned short f2bf_raw(float f) {
  union { float f; unsigned u; } a; a.f = f;
  unsigned r = a.u + 0x7FFFu + ((a.u >> 16) & 1u);  // round-to-nearest-even
  return (unsigned short)(r >> 16);
}

__device__ __forceinline__ v8f wmma_bf16(v16bf a, v16bf b, v8f c) {
  // D = A(16x32 bf16) x B(32x16 bf16) + C(16x16 f32)
  return __builtin_amdgcn_wmma_f32_16x16x32_bf16(false, a, false, b, (short)0, c,
                                                 false, false);
}

// ---------------------------------------------------------------------------
// Fragment loaders per CDNA5 ISA 7.12.2 (wave32).
// A (16x32, 16-bit): lane 0-15 = rows; a lane's 16 elements are two contiguous
// 8-element K runs: {half*8 .. half*8+7} and {16+half*8 .. 16+half*8+7}.
//   -> exactly two B128 loads.
// ---------------------------------------------------------------------------
__device__ __forceinline__ v16bf load_a_bf16(const unsigned short* A, int m0,
                                             int k0, int lda) {
  const int lane = threadIdx.x & 31;
  const int half = lane >> 4, r = lane & 15;
  const unsigned short* row = A + (size_t)(m0 + r) * lda + k0 + (half << 3);
  union { v16bf v; uint4 q[2]; } u;
  u.q[0] = *(const uint4*)(row);        // K = half*8 + 0..7
  u.q[1] = *(const uint4*)(row + 16);   // K = 16 + half*8 + 0..7
  return u.v;
}

// B (32x16, 16-bit): lane 0-15 = cols; a lane's 16 elements are two K runs
// {half*8 .. half*8+7} and {16+half*8 .. 16+half*8+7} (same as A), with the
// source storing the B operand transposed, element (k,n) at S[n*ld + k].  -> two B128 loads.
__device__ __forceinline__ v16bf load_bt_bf16(const unsigned short* S, int k0,
                                              int n0, int ld) {
  const int lane = threadIdx.x & 31;
  const int half = lane >> 4, c = lane & 15;
  const unsigned short* col = S + (size_t)(n0 + c) * ld + k0 + (half << 3);
  union { v16bf v; uint4 q[2]; } u;
  u.q[0] = *(const uint4*)(col);        // K = half*8 + 0..7
  u.q[1] = *(const uint4*)(col + 16);   // K = 16 + half*8 + 0..7
  return u.v;
}

// ---------------------------------------------------------------------------
// 0) one-shot f32 -> bf16 conversion (weights)
// ---------------------------------------------------------------------------
__global__ void convert_kernel(const float* __restrict__ src,
                               unsigned short* __restrict__ dst, int n) {
  int i = blockIdx.x * blockDim.x + threadIdx.x;
  if (i < n) dst[i] = f2bf_raw(src[i]);
}

// ---------------------------------------------------------------------------
// 1) transpose + positional encoding: x(B,C,L) -> t(B,L,C) fp32 + bf16
// ---------------------------------------------------------------------------
__global__ void prep_kernel(const float* __restrict__ x, float* __restrict__ t,
                            unsigned short* __restrict__ tbf) {
  int idx = blockIdx.x * blockDim.x + threadIdx.x;
  if (idx >= B_ * L_ * C_) return;
  int c = idx & (C_ - 1);
  int l = (idx / C_) & (L_ - 1);
  int b = idx / (C_ * L_);
  const float inv = -logf(10000.0f) / (float)C_;
  float ang = (float)l * __expf((float)((c >> 1) << 1) * inv);
  float pe = (c & 1) ? cosf(ang) : sinf(ang);
  float v = x[(size_t)b * C_ * L_ + (size_t)c * L_ + l] + pe;
  t[idx] = v;
  tbf[idx] = f2bf_raw(v);
}

// ---------------------------------------------------------------------------
// 2) QKV projection GEMM, 32x32 tile per wave (4 WMMAs / K-step).
//    which==2 (V) is stored transposed (B,H,D,L) for the P@V operand.
// ---------------------------------------------------------------------------
__global__ void qkv_kernel(const unsigned short* __restrict__ tbf,
                           const unsigned short* __restrict__ wq,
                           const unsigned short* __restrict__ wk,
                           const unsigned short* __restrict__ wv,
                           const float* __restrict__ bq,
                           const float* __restrict__ bk,
                           const float* __restrict__ bv,
                           unsigned short* __restrict__ qo,
                           unsigned short* __restrict__ ko,
                           unsigned short* __restrict__ vt) {
  const int m0 = blockIdx.x << 5, n0 = blockIdx.y << 5, which = blockIdx.z;
  const unsigned short* W = (which == 0) ? wq : (which == 1) ? wk : wv;
  const float* bias = (which == 0) ? bq : (which == 1) ? bk : bv;
  const float sc = (which == 0) ? 0.125f : 1.0f;  // 1/sqrt(D)

  v8f acc00 = {0.f,0.f,0.f,0.f,0.f,0.f,0.f,0.f}, acc01 = acc00;
  v8f acc10 = acc00, acc11 = acc00;
#pragma unroll
  for (int k0 = 0; k0 < C_; k0 += 32) {
    if (k0 + 32 < C_) {
      __builtin_prefetch(tbf + (size_t)(m0 + (threadIdx.x & 15)) * C_ + k0 + 32, 0, 1);
    }
    v16bf a0 = load_a_bf16(tbf, m0, k0, C_);
    v16bf a1 = load_a_bf16(tbf, m0 + 16, k0, C_);
    v16bf b0 = load_bt_bf16(W, k0, n0, C_);       // W row-major (HD,C) == B^T
    v16bf b1 = load_bt_bf16(W, k0, n0 + 16, C_);
    acc00 = wmma_bf16(a0, b0, acc00);
    acc01 = wmma_bf16(a0, b1, acc01);
    acc10 = wmma_bf16(a1, b0, acc10);
    acc11 = wmma_bf16(a1, b1, acc11);
  }
  const int lane = threadIdx.x & 31, half = lane >> 4, c = lane & 15;
  const v8f* accs[4] = {&acc00, &acc01, &acc10, &acc11};
#pragma unroll
  for (int mi = 0; mi < 2; ++mi) {
#pragma unroll
    for (int ni = 0; ni < 2; ++ni) {
      const v8f& acc = *accs[mi * 2 + ni];
      int col = n0 + (ni << 4) + c;
      float bv_ = bias[col];
#pragma unroll
      for (int i = 0; i < 8; ++i) {
        int row = m0 + (mi << 4) + i + (half << 3);
        unsigned short val = f2bf_raw((acc[i] + bv_) * sc);
        if (which == 2) {
          int bb = row / L_, l = row & (L_ - 1);
          int h = col >> 6, d = col & (D_ - 1);
          vt[(size_t)((bb * H_ + h) * D_ + d) * L_ + l] = val;
        } else {
          unsigned short* out = (which == 0) ? qo : ko;
          out[(size_t)row * HD_ + col] = val;
        }
      }
    }
  }
}

// ---------------------------------------------------------------------------
// 3) Flash attention: block = 4 waves x 16 query rows; stream K/V in 32-chunks
//    All operand loads are B128 (Q/P: A-layout runs; K^T, V^T: K-contiguous).
// ---------------------------------------------------------------------------
__global__ void attn_kernel(const unsigned short* __restrict__ q,
                            const unsigned short* __restrict__ kk,
                            const unsigned short* __restrict__ vt,
                            unsigned short* __restrict__ o) {
  __shared__ __align__(16) unsigned short ldsP[4][16 * 32];  // per-wave P tile
  const int wave = threadIdx.x >> 5;
  const int lane = threadIdx.x & 31;
  const int half = lane >> 4, lc = lane & 15;
  const int bh = blockIdx.y, b = bh / H_, h = bh % H_;
  const int m0 = (blockIdx.x << 6) + (wave << 4);
  const unsigned short* Q  = q  + (size_t)b * L_ * HD_ + h * D_;  // (l,d) ld=HD
  const unsigned short* K  = kk + (size_t)b * L_ * HD_ + h * D_;  // (s,d) ld=HD
  const unsigned short* Vt = vt + (size_t)(b * H_ + h) * D_ * L_; // (d,s) ld=L

  v16bf aQ0 = load_a_bf16(Q, m0, 0, HD_);
  v16bf aQ1 = load_a_bf16(Q, m0, 32, HD_);

  float mrow[8], lrow[8];
  v8f accO[4];
#pragma unroll
  for (int i = 0; i < 8; ++i) { mrow[i] = -1e30f; lrow[i] = 0.f; }
#pragma unroll
  for (int j = 0; j < 4; ++j) accO[j] = (v8f){0.f,0.f,0.f,0.f,0.f,0.f,0.f,0.f};

  for (int s0 = 0; s0 < L_; s0 += 32) {
    // S = Q K^T : two 16x16 subtiles over D=64 (2 WMMAs each)
    v8f sa = {0.f,0.f,0.f,0.f,0.f,0.f,0.f,0.f};
    v8f sb = sa;
    {
      v16bf b0 = load_bt_bf16(K, 0, s0, HD_);      // K^T(d,s) = K[s*HD+d]
      v16bf b1 = load_bt_bf16(K, 32, s0, HD_);
      sa = wmma_bf16(aQ0, b0, sa);
      sa = wmma_bf16(aQ1, b1, sa);
      v16bf b2 = load_bt_bf16(K, 0, s0 + 16, HD_);
      v16bf b3 = load_bt_bf16(K, 32, s0 + 16, HD_);
      sb = wmma_bf16(aQ0, b2, sb);
      sb = wmma_bf16(aQ1, b3, sb);
    }
    // online softmax (row reductions across the 16-lane half via shfl_xor)
    float p0[8], p1[8];
#pragma unroll
    for (int i = 0; i < 8; ++i) {
      float rm = fmaxf(sa[i], sb[i]);
#pragma unroll
      for (int msk = 1; msk < 16; msk <<= 1)
        rm = fmaxf(rm, __shfl_xor(rm, msk, 32));
      float mnew = fmaxf(mrow[i], rm);
      float esc = __expf(mrow[i] - mnew);
      float e0 = __expf(sa[i] - mnew);
      float e1 = __expf(sb[i] - mnew);
      float rs = e0 + e1;
#pragma unroll
      for (int msk = 1; msk < 16; msk <<= 1) rs += __shfl_xor(rs, msk, 32);
      lrow[i] = lrow[i] * esc + rs;
      mrow[i] = mnew;
      p0[i] = e0; p1[i] = e1;
#pragma unroll
      for (int j = 0; j < 4; ++j) accO[j][i] = accO[j][i] * esc;
    }
    // stage P (C-layout) into per-wave LDS, reload as A-layout fragment
    unsigned short* P = &ldsP[wave][0];
#pragma unroll
    for (int i = 0; i < 8; ++i) {
      int row = i + (half << 3);
      P[row * 32 + lc] = f2bf_raw(p0[i]);
      P[row * 32 + 16 + lc] = f2bf_raw(p1[i]);
    }
    __syncthreads();  // uniform; orders ds_store -> ds_load
    v16bf aP = load_a_bf16(P, 0, 0, 32);           // 2x ds_load_b128
    // O += P x V : V^T(d,s) K-contiguous -> B128 loads
#pragma unroll
    for (int j = 0; j < 4; ++j) {
      v16bf bV = load_bt_bf16(Vt, s0, j << 4, L_);
      accO[j] = wmma_bf16(aP, bV, accO[j]);
    }
  }
  // normalize and emit o in (B,L,HD) bf16
#pragma unroll
  for (int i = 0; i < 8; ++i) {
    float inv = 1.0f / lrow[i];
    int row = m0 + i + (half << 3);
#pragma unroll
    for (int j = 0; j < 4; ++j) {
      o[(size_t)(b * L_ + row) * HD_ + h * D_ + (j << 4) + lc] =
          f2bf_raw(accO[j][i] * inv);
    }
  }
}

// ---------------------------------------------------------------------------
// 4) output projection + residual: t2 = t + o@Wo^T + bo  (M=4096,K=512,N=256)
//    32x32 tile per wave.
// ---------------------------------------------------------------------------
__global__ void outproj_kernel(const unsigned short* __restrict__ ob,
                               const unsigned short* __restrict__ wo,
                               const float* __restrict__ bo,
                               const float* __restrict__ t,
                               float* __restrict__ t2) {
  const int m0 = blockIdx.x << 5, n0 = blockIdx.y << 5;
  v8f acc00 = {0.f,0.f,0.f,0.f,0.f,0.f,0.f,0.f}, acc01 = acc00;
  v8f acc10 = acc00, acc11 = acc00;
#pragma unroll
  for (int k0 = 0; k0 < HD_; k0 += 32) {
    if (k0 + 32 < HD_) {
      __builtin_prefetch(ob + (size_t)(m0 + (threadIdx.x & 15)) * HD_ + k0 + 32, 0, 1);
    }
    v16bf a0 = load_a_bf16(ob, m0, k0, HD_);
    v16bf a1 = load_a_bf16(ob, m0 + 16, k0, HD_);
    v16bf b0 = load_bt_bf16(wo, k0, n0, HD_);      // Wo (C,HD) == B^T
    v16bf b1 = load_bt_bf16(wo, k0, n0 + 16, HD_);
    acc00 = wmma_bf16(a0, b0, acc00);
    acc01 = wmma_bf16(a0, b1, acc01);
    acc10 = wmma_bf16(a1, b0, acc10);
    acc11 = wmma_bf16(a1, b1, acc11);
  }
  const int lane = threadIdx.x & 31, half = lane >> 4, c = lane & 15;
  const v8f* accs[4] = {&acc00, &acc01, &acc10, &acc11};
#pragma unroll
  for (int mi = 0; mi < 2; ++mi) {
#pragma unroll
    for (int ni = 0; ni < 2; ++ni) {
      const v8f& acc = *accs[mi * 2 + ni];
      int col = n0 + (ni << 4) + c;
      float bv_ = bo[col];
#pragma unroll
      for (int i = 0; i < 8; ++i) {
        size_t idx = (size_t)(m0 + (mi << 4) + i + (half << 3)) * C_ + col;
        t2[idx] = t[idx] + acc[i] + bv_;
      }
    }
  }
}

// ---------------------------------------------------------------------------
// 5) FFN conv GEMM per batch: f = conv_w(C,C) @ y(C,L) + conv_b, 32x32 tiles.
//    y_bf is stored transposed (B,L-major? no): y is (B,C,L); B operand (k=c,
//    n=l) = y[c*L + l] is N-contiguous, so we store y ALSO transposed as
//    (B,L,C) bf16 (yt) making the operand K-contiguous.
// ---------------------------------------------------------------------------
__global__ void conv_kernel(const unsigned short* __restrict__ cwbf,
                            const float* __restrict__ cb,
                            const unsigned short* __restrict__ yt,
                            float* __restrict__ f) {
  const int m0 = blockIdx.x << 5, n0 = blockIdx.y << 5, b = blockIdx.z;
  const unsigned short* Bt = yt + (size_t)b * C_ * L_;  // (l,c): yt[l*C + c]
  v8f acc00 = {0.f,0.f,0.f,0.f,0.f,0.f,0.f,0.f}, acc01 = acc00;
  v8f acc10 = acc00, acc11 = acc00;
#pragma unroll
  for (int k0 = 0; k0 < C_; k0 += 32) {
    v16bf a0 = load_a_bf16(cwbf, m0, k0, C_);       // conv_w row-major (C,C)
    v16bf a1 = load_a_bf16(cwbf, m0 + 16, k0, C_);
    v16bf b0 = load_bt_bf16(Bt, k0, n0, C_);        // B^T(n=l, k=c) K-contig
    v16bf b1 = load_bt_bf16(Bt, k0, n0 + 16, C_);
    acc00 = wmma_bf16(a0, b0, acc00);
    acc01 = wmma_bf16(a0, b1, acc01);
    acc10 = wmma_bf16(a1, b0, acc10);
    acc11 = wmma_bf16(a1, b1, acc11);
  }
  const int lane = threadIdx.x & 31, half = lane >> 4, c = lane & 15;
  const v8f* accs[4] = {&acc00, &acc01, &acc10, &acc11};
#pragma unroll
  for (int mi = 0; mi < 2; ++mi) {
#pragma unroll
    for (int ni = 0; ni < 2; ++ni) {
      const v8f& acc = *accs[mi * 2 + ni];
      int col = n0 + (ni << 4) + c;  // l
#pragma unroll
      for (int i = 0; i < 8; ++i) {
        int row = m0 + (mi << 4) + i + (half << 3);  // output channel
        f[(size_t)b * C_ * L_ + (size_t)row * L_ + col] = acc[i] + cb[row];
      }
    }
  }
}

// ---------------------------------------------------------------------------
// Global-LN stats helpers (per batch over C*L elements)
// ---------------------------------------------------------------------------
__global__ void zero_stats_kernel(float* stats) {
  if (threadIdx.x < 12) stats[threadIdx.x] = 0.f;
}

__device__ __forceinline__ void block_reduce_atomic(float s, float ss,
                                                    float* sum_p, float* ssq_p) {
#pragma unroll
  for (int msk = 1; msk < 32; msk <<= 1) {
    s += __shfl_xor(s, msk, 32);
    ss += __shfl_xor(ss, msk, 32);
  }
  __shared__ float sh[2][8];
  int wave = threadIdx.x >> 5, lane = threadIdx.x & 31;
  if (lane == 0) { sh[0][wave] = s; sh[1][wave] = ss; }
  __syncthreads();
  if (threadIdx.x == 0) {
    float ts = 0.f, tss = 0.f;
    int nw = (int)(blockDim.x >> 5);
    for (int w = 0; w < nw; ++w) { ts += sh[0][w]; tss += sh[1][w]; }
    atomicAdd(sum_p, ts);
    atomicAdd(ssq_p, tss);
  }
}

__global__ void stats_kernel(const float* __restrict__ src,
                             float* __restrict__ stats, int nper) {
  const int b = blockIdx.y;
  float s = 0.f, ss = 0.f;
  for (int i = blockIdx.x * blockDim.x + threadIdx.x; i < nper;
       i += gridDim.x * blockDim.x) {
    float v = src[(size_t)b * nper + i];
    s += v; ss += v * v;
  }
  block_reduce_atomic(s, ss, &stats[b * 2], &stats[b * 2 + 1]);
}

// 6) LN after attention: t2(B,L,C) -> y(B,C,L) fp32 and yt(B,L,C) bf16
__global__ void ln_mha_kernel(const float* __restrict__ t2,
                              const float* __restrict__ g,
                              const float* __restrict__ be,
                              const float* __restrict__ stats,
                              float* __restrict__ y,
                              unsigned short* __restrict__ yt) {
  int idx = blockIdx.x * blockDim.x + threadIdx.x;
  if (idx >= B_ * L_ * C_) return;
  int c = idx & (C_ - 1);
  int l = (idx / C_) & (L_ - 1);
  int b = idx / (C_ * L_);
  const float n = (float)(C_ * L_);
  float mean = stats[b * 2] / n;
  float var = stats[b * 2 + 1] / n - mean * mean;
  float rstd = rsqrtf(var + EPS_);
  float v = g[c] * ((t2[idx] - mean) * rstd) + be[c];
  y[(size_t)b * C_ * L_ + (size_t)c * L_ + l] = v;  // (B,C,L)
  yt[idx] = f2bf_raw(v);                            // (B,L,C) for conv B^T
}

// 7) fused: LN(f) -> PReLU -> +y  (in-place into f), accumulates final stats
__global__ void ffn_mid_kernel(float* __restrict__ f,
                               const float* __restrict__ y,
                               const float* __restrict__ g,
                               const float* __restrict__ be,
                               const float* __restrict__ alpha_p,
                               float* __restrict__ stats) {
  const int b = blockIdx.y;
  const float n = (float)(C_ * L_);
  float mean = stats[4 + b * 2] / n;
  float var = stats[5 + b * 2] / n - mean * mean;
  float rstd = rsqrtf(var + EPS_);
  float alpha = alpha_p[0];
  float s = 0.f, ss = 0.f;
  for (int i = blockIdx.x * blockDim.x + threadIdx.x; i < C_ * L_;
       i += gridDim.x * blockDim.x) {
    int c = i / L_;
    size_t idx = (size_t)b * C_ * L_ + i;
    float fv = g[c] * ((f[idx] - mean) * rstd) + be[c];
    fv = (fv >= 0.f) ? fv : alpha * fv;
    float z = fv + y[idx];
    f[idx] = z;
    s += z; ss += z * z;
  }
  block_reduce_atomic(s, ss, &stats[8 + b * 2], &stats[9 + b * 2]);
}

// 8) final LN -> d_out (B,C,L) fp32
__global__ void final_kernel(const float* __restrict__ z,
                             const float* __restrict__ g,
                             const float* __restrict__ be,
                             const float* __restrict__ stats,
                             float* __restrict__ out) {
  int idx = blockIdx.x * blockDim.x + threadIdx.x;
  if (idx >= B_ * C_ * L_) return;
  int c = (idx / L_) & (C_ - 1);
  int b = idx / (C_ * L_);
  const float n = (float)(C_ * L_);
  float mean = stats[8 + b * 2] / n;
  float var = stats[9 + b * 2] / n - mean * mean;
  float rstd = rsqrtf(var + EPS_);
  out[idx] = g[c] * ((z[idx] - mean) * rstd) + be[c];
}

// ---------------------------------------------------------------------------
extern "C" void kernel_launch(void* const* d_in, const int* in_sizes, int n_in,
                              void* d_out, int out_size, void* d_ws,
                              size_t ws_size, hipStream_t stream) {
  const float* x     = (const float*)d_in[0];
  const float* Wq    = (const float*)d_in[1];
  const float* bq    = (const float*)d_in[2];
  const float* Wk    = (const float*)d_in[3];
  const float* bk    = (const float*)d_in[4];
  const float* Wv    = (const float*)d_in[5];
  const float* bv    = (const float*)d_in[6];
  const float* Wo    = (const float*)d_in[7];
  const float* bo    = (const float*)d_in[8];
  const float* g_mha = (const float*)d_in[9];
  const float* b_mha = (const float*)d_in[10];
  const float* cw    = (const float*)d_in[11];
  const float* cb    = (const float*)d_in[12];
  const float* g_ffn = (const float*)d_in[13];
  const float* b_ffn = (const float*)d_in[14];
  const float* alpha = (const float*)d_in[15];
  const float* g_out = (const float*)d_in[16];
  const float* b_out = (const float*)d_in[17];

  char* ws = (char*)d_ws;
  const size_t MB = 1024u * 1024u;
  const size_t KB = 1024u;
  float*          t_f  = (float*)(ws + 0 * MB);           // (B,L,C) fp32  4MB
  unsigned short* t_bf = (unsigned short*)(ws + 4 * MB);  // (B,L,C) bf16  2MB
  unsigned short* q_bf = (unsigned short*)(ws + 6 * MB);  // (B,L,HD)      4MB
  unsigned short* k_bf = (unsigned short*)(ws + 10 * MB); // (B,L,HD)      4MB
  unsigned short* v_t  = (unsigned short*)(ws + 14 * MB); // (B,H,D,L)     4MB
  unsigned short* o_bf = (unsigned short*)(ws + 18 * MB); // (B,L,HD)      4MB
  float*          t2   = (float*)(ws + 22 * MB);          // (B,L,C) fp32  4MB
  float*          y_f  = (float*)(ws + 26 * MB);          // (B,C,L) fp32  4MB
  unsigned short* y_t  = (unsigned short*)(ws + 30 * MB); // (B,L,C) bf16  2MB
  float*          f_f  = (float*)(ws + 32 * MB);          // (B,C,L) fp32  4MB
  float*          st   = (float*)(ws + 36 * MB);          // 12 floats
  unsigned short* wq_bf = (unsigned short*)(ws + 37 * MB);            // 256KB
  unsigned short* wk_bf = (unsigned short*)(ws + 37 * MB + 256 * KB); // 256KB
  unsigned short* wv_bf = (unsigned short*)(ws + 37 * MB + 512 * KB); // 256KB
  unsigned short* wo_bf = (unsigned short*)(ws + 37 * MB + 768 * KB); // 256KB
  unsigned short* cw_bf = (unsigned short*)(ws + 38 * MB);            // 128KB

  const int NE = B_ * L_ * C_;
  const int NW = HD_ * C_;  // 131072

  // one-shot weight conversions
  convert_kernel<<<(NW + 255) / 256, 256, 0, stream>>>(Wq, wq_bf, NW);
  convert_kernel<<<(NW + 255) / 256, 256, 0, stream>>>(Wk, wk_bf, NW);
  convert_kernel<<<(NW + 255) / 256, 256, 0, stream>>>(Wv, wv_bf, NW);
  convert_kernel<<<(NW + 255) / 256, 256, 0, stream>>>(Wo, wo_bf, NW);
  convert_kernel<<<(C_ * C_ + 255) / 256, 256, 0, stream>>>(cw, cw_bf, C_ * C_);

  prep_kernel<<<(NE + 255) / 256, 256, 0, stream>>>(x, t_f, t_bf);
  zero_stats_kernel<<<1, 32, 0, stream>>>(st);
  qkv_kernel<<<dim3((B_ * L_) / 32, HD_ / 32, 3), 32, 0, stream>>>(
      t_bf, wq_bf, wk_bf, wv_bf, bq, bk, bv, q_bf, k_bf, v_t);
  attn_kernel<<<dim3(L_ / 64, B_ * H_), 128, 0, stream>>>(q_bf, k_bf, v_t,
                                                          o_bf);
  outproj_kernel<<<dim3((B_ * L_) / 32, C_ / 32), 32, 0, stream>>>(
      o_bf, wo_bf, bo, t_f, t2);
  stats_kernel<<<dim3(64, B_), 256, 0, stream>>>(t2, st + 0, C_ * L_);
  ln_mha_kernel<<<(NE + 255) / 256, 256, 0, stream>>>(t2, g_mha, b_mha, st,
                                                      y_f, y_t);
  conv_kernel<<<dim3(C_ / 32, L_ / 32, B_), 32, 0, stream>>>(cw_bf, cb, y_t,
                                                             f_f);
  stats_kernel<<<dim3(64, B_), 256, 0, stream>>>(f_f, st + 4, C_ * L_);
  ffn_mid_kernel<<<dim3(64, B_), 256, 0, stream>>>(f_f, y_f, g_ffn, b_ffn,
                                                   alpha, st);
  final_kernel<<<(NE + 255) / 256, 256, 0, stream>>>(f_f, g_out, b_out, st,
                                                     (float*)d_out);
}
